// SDEBase_50448685858834
// MI455X (gfx1250) — matrix operand order repaired and checked by reading, not compiled, not run on hardware
//
#include <hip/hip_runtime.h>
#include <hip/hip_bf16.h>
#include <math.h>

typedef __attribute__((ext_vector_type(16))) _Float16 v16h;
typedef __attribute__((ext_vector_type(8)))  float    v8f;

#define NB      16384
#define NS      16
#define NSTEPS  50
#define LATENT  128
#define HID     64
#define DT_F    0.02f
#define SQRT_DT 0.14142135623730951f

// A-matrix 16x32: lane holds row m=lane%16; k = (lane>>4)*16 + hidx
__device__ __forceinline__ int a_klocal(int hidx, int hi) {
    return hi * 16 + hidx;
}
// B-matrix 32x16: lane holds col n=lane%16; k = (lane>>4)*16 + hidx
__device__ __forceinline__ int b_klocal(int hidx, int hi) {
    return hi * 16 + hidx;
}

// fast device math: native v_exp/v_log/v_rcp trans ops (results feed f16
// WMMA fragments, so OCML-precise versions are wasted latency)
__device__ __forceinline__ float frcp(float x)  { return __builtin_amdgcn_rcpf(x); }
__device__ __forceinline__ float fast_tanhf(float x) {
    // 1 - 2/(e^{2x}+1); saturates correctly to +/-1 for large |x|
    return 1.f - 2.f * frcp(__expf(2.f * x) + 1.f);
}
__device__ __forceinline__ float fast_softplusf(float x) {
    return (x > 20.f) ? x : __logf(1.f + __expf(x));
}
__device__ __forceinline__ float softplusf(float x) {   // precise (epilogue)
    return (x > 20.f) ? x : log1pf(expf(x));
}
__device__ __forceinline__ float fixnum(float v) {
    if (v != v) return 0.f;
    if (v >  3.402823466e38f) return  3.402823466e38f;
    if (v < -3.402823466e38f) return -3.402823466e38f;
    return v;
}

// ---------------------------------------------------------------------------
// Kernel 1: base[b][j] = z[b]·W1[j,2:] + b1[j]  (B x 64 via WMMA, K=128),
// boundary/ndt heads folded in as N columns 64 (Wb) and 65 (Wn).
// One wave (= one workgroup) per 16-row tile; N=80 -> 5 n-tiles; K=128 ->
// 4 k-steps; 20 WMMAs per wave. All fragment loads are unconditional,
// contiguous 64B runs per lane (merge into global_load_b128).
// ---------------------------------------------------------------------------
__global__ __launch_bounds__(32) void sde_base_gemm(
    const float* __restrict__ z,  const float* __restrict__ W1,
    const float* __restrict__ b1, const float* __restrict__ Wb,
    const float* __restrict__ bb, const float* __restrict__ Wn,
    const float* __restrict__ bn,
    float* __restrict__ base_out, float* __restrict__ bnd_out,
    float* __restrict__ ndt_out)
{
    const int lane = threadIdx.x & 31;
    const int b0   = blockIdx.x * 16;
    const int m    = lane & 15;
    const int hi   = lane >> 4;

    // head tile (t=4) column source, selected once per lane (branchless)
    const float* col4 = (m == 1) ? Wn : Wb;
    const float  msk4 = (m < 2) ? 1.f : 0.f;

    v8f acc[5] = {};

    #pragma unroll
    for (int kk = 0; kk < 4; ++kk) {
        const int K0 = kk * 32;
        v16h afrag;
        #pragma unroll
        for (int hidx = 0; hidx < 16; ++hidx) {
            int k = K0 + a_klocal(hidx, hi);
            afrag[hidx] = (_Float16)z[(size_t)(b0 + m) * LATENT + k];
        }
        // n-tiles 0..3: always a W1 row -> unconditional contiguous loads
        #pragma unroll
        for (int t = 0; t < 4; ++t) {
            const float* colp = W1 + (size_t)(t * 16 + m) * (LATENT + 2) + 2;
            v16h bfrag;
            #pragma unroll
            for (int hidx = 0; hidx < 16; ++hidx)
                bfrag[hidx] = (_Float16)colp[K0 + b_klocal(hidx, hi)];
            acc[t] = __builtin_amdgcn_wmma_f32_16x16x32_f16(
                false, afrag, false, bfrag, (short)0, acc[t], false, false);
        }
        // head tile: col 0 = Wb, col 1 = Wn, cols 2..15 zero-masked
        {
            v16h bfrag;
            #pragma unroll
            for (int hidx = 0; hidx < 16; ++hidx)
                bfrag[hidx] = (_Float16)(col4[K0 + b_klocal(hidx, hi)] * msk4);
            acc[4] = __builtin_amdgcn_wmma_f32_16x16x32_f16(
                false, afrag, false, bfrag, (short)0, acc[4], false, false);
        }
    }

    // epilogue: D layout -> row = r + 8*hi, col = ntile*16 + m
    #pragma unroll
    for (int t = 0; t < 4; ++t) {
        const int   n    = t * 16 + m;
        const float bias = b1[n];
        #pragma unroll
        for (int r = 0; r < 8; ++r) {
            int row = b0 + r + 8 * hi;
            base_out[(size_t)row * HID + n] = acc[t][r] + bias;
        }
    }
    if (m < 2) {   // head columns: boundary (m==0) / ndt (m==1)
        const float  hb  = (m == 0) ? bb[0]  : bn[0];
        const float  ha  = (m == 0) ? 0.3f   : 0.05f;
        float*       dst = (m == 0) ? bnd_out : ndt_out;
        #pragma unroll
        for (int r = 0; r < 8; ++r) {
            int row = b0 + r + 8 * hi;
            dst[row] = softplusf(acc[4][r] + hb) + ha;
        }
    }
}

// ---------------------------------------------------------------------------
// Kernel 2: 50-step SDE. One wave (= one single-wave workgroup) per batch
// element b; the 16 sims are the 16 rows of a WMMA tile. Per step: fast-tanh
// activations (v_exp/v_rcp trans ops) built directly in the A-fragment
// layout, two v_wmma_f32_16x16x32_f16 against a zero-padded W2 B-fragment
// (col0=drift, col1=diff), intra-wave LDS scatter of the 2 live columns
// (LDS ops of one wave are in-order -> wave_barrier only, no s_barrier),
// then f32 pointwise hazard/survival updates. State is replicated in both
// half-waves so no cross-lane x broadcast is needed.
// ---------------------------------------------------------------------------
__global__ __launch_bounds__(32) void sde_steps(
    const float* __restrict__ W1,      const float* __restrict__ W2,
    const float* __restrict__ b2,      const float* __restrict__ noise,
    const float* __restrict__ base_ws, const float* __restrict__ bnd_ws,
    const float* __restrict__ ndt_ws,  const float* __restrict__ out_scale,
    const float* __restrict__ out_bias, float* __restrict__ out)
{
    __shared__ float lds[32];

    const int lane = threadIdx.x & 31;
    const int b    = blockIdx.x;
    const int m    = lane & 15;   // sim index
    const int hi   = lane >> 4;

    // Per-lane hidden-unit slice (32 of 64 k's, fixed by A-fragment layout)
    float bbuf[32], w1x[32], w1tdt[32];
    #pragma unroll
    for (int i = 0; i < 32; ++i) {
        int k = ((i >> 4) * 32) + a_klocal(i & 15, hi);
        bbuf[i]  = base_ws[(size_t)b * HID + k];      // z·W1 + b1 (t=0 term)
        w1x[i]   = W1[k * (LATENT + 2) + 0];          // x column
        w1tdt[i] = W1[k * (LATENT + 2) + 1] * DT_F;   // t column, per-step inc
    }

    // Constant B fragments: W2 padded to 32x16 (only n=0,1 nonzero)
    v16h Bf0, Bf1;
    {
        const float* rowp = W2 + (size_t)m * HID;     // valid for m<2
        const float  wmsk = (m < 2) ? 1.f : 0.f;
        #pragma unroll
        for (int hidx = 0; hidx < 16; ++hidx) {
            int k0 = b_klocal(hidx, hi);              // 0..31
            Bf0[hidx] = (_Float16)(rowp[k0]      * wmsk);
            Bf1[hidx] = (_Float16)(rowp[32 + k0] * wmsk);
        }
    }

    const float half_b = 0.5f * bnd_ws[b];
    const float b20 = b2[0], b21 = b2[1];

    float x = 0.f, log_surv = 0.f, exp_rt = 0.f, exp_corr = 0.f;

    for (int step = 0; step < NSTEPS; ++step) {
        // prefetch next step's noise line (global_prefetch path)
        if (step + 1 < NSTEPS)
            __builtin_prefetch(
                &noise[(size_t)(step + 1) * NB * NS + (size_t)b * NS + m], 0, 1);

        // h = tanh(x*w1x + (z·W1 + b1 + t*w1t)) packed straight into A frags
        v16h A0, A1;
        #pragma unroll
        for (int hidx = 0; hidx < 16; ++hidx) {
            float p0 = fmaf(x, w1x[hidx],      bbuf[hidx]);
            float p1 = fmaf(x, w1x[16 + hidx], bbuf[16 + hidx]);
            A0[hidx] = (_Float16)fast_tanhf(p0);
            A1[hidx] = (_Float16)fast_tanhf(p1);
        }
        #pragma unroll
        for (int i = 0; i < 32; ++i) bbuf[i] += w1tdt[i];   // advance t term

        v8f acc = {};
        acc = __builtin_amdgcn_wmma_f32_16x16x32_f16(
            false, A0, false, Bf0, (short)0, acc, false, false);
        acc = __builtin_amdgcn_wmma_f32_16x16x32_f16(
            false, A1, false, Bf1, (short)0, acc, false, false);

        // scatter the two live output columns (drift/diff) per sim via LDS;
        // same-wave LDS ops are in-order (ISA 7.3), so only a compile-time
        // ordering fence is required
        if (m < 2) {
            #pragma unroll
            for (int r = 0; r < 8; ++r)
                lds[m * 16 + r + 8 * hi] = acc[r];
        }
        __builtin_amdgcn_wave_barrier();
        float drift = fminf(fmaxf(lds[m] + b20, -5.f), 5.f);
        float diff  = fast_softplusf(lds[16 + m] + b21) + 0.1f;
        __builtin_amdgcn_wave_barrier();

        float nz = noise[(size_t)step * NB * NS + (size_t)b * NS + m];
        x = fminf(fmaxf(x + drift * DT_F + diff * SQRT_DT * nz, -10.f), 10.f);

        float dist   = fabsf(x) - half_b;
        float hazard = fminf(frcp(1.f + __expf(-20.f * dist)), 0.99f);
        float surv_b = __expf(fmaxf(log_surv, -50.f));
        float crossp = surv_b * hazard;
        log_surv += __logf(1.f - hazard);          // hazard <= 0.99, safe
        float t_now = (float)(step + 1) * DT_F;
        exp_rt   += crossp * t_now;
        exp_corr += crossp * (x > 0.f ? 1.f : 0.f);
    }

    float remaining = expf(fmaxf(log_surv, -50.f));
    exp_rt   += remaining * 1.0f;            // STEPS * DT = 1.0
    exp_corr += remaining * 0.5f;
    float rt_ms = (exp_rt + ndt_ws[b]) * 1000.f;

    // mean / unbiased std / corr mean over the 16 sims (lanes 0..15 and
    // 16..31 hold identical replicas, so xor-reduce within each 16-group)
    float s1 = rt_ms, s2 = rt_ms * rt_ms, s3 = exp_corr;
    #pragma unroll
    for (int mask = 8; mask >= 1; mask >>= 1) {
        s1 += __shfl_xor(s1, mask, 32);
        s2 += __shfl_xor(s2, mask, 32);
        s3 += __shfl_xor(s3, mask, 32);
    }
    if (lane == 0) {
        float mean = s1 * (1.f / 16.f);
        float var  = (s2 - 16.f * mean * mean) * (1.f / 15.f);
        float stdv = sqrtf(fmaxf(var, 0.f)) + 0.001f;
        float cm   = s3 * (1.f / 16.f);
        out[b * 3 + 0] = fixnum(fmaf(mean, out_scale[0], out_bias[0]));
        out[b * 3 + 1] = fixnum(fmaf(stdv, out_scale[1], out_bias[1]));
        out[b * 3 + 2] = fixnum(fmaf(cm,   out_scale[2], out_bias[2]));
    }
}

extern "C" void kernel_launch(void* const* d_in, const int* in_sizes, int n_in,
                              void* d_out, int out_size, void* d_ws, size_t ws_size,
                              hipStream_t stream) {
    (void)in_sizes; (void)n_in; (void)out_size; (void)ws_size;
    const float* z         = (const float*)d_in[0];
    const float* W1        = (const float*)d_in[1];
    const float* b1        = (const float*)d_in[2];
    const float* W2        = (const float*)d_in[3];
    const float* b2        = (const float*)d_in[4];
    const float* Wb        = (const float*)d_in[5];
    const float* bb        = (const float*)d_in[6];
    const float* Wn        = (const float*)d_in[7];
    const float* bn        = (const float*)d_in[8];
    const float* out_scale = (const float*)d_in[9];
    const float* out_bias  = (const float*)d_in[10];
    const float* noise     = (const float*)d_in[11];
    float* out = (float*)d_out;

    float* base_ws = (float*)d_ws;                 // NB*64 f32 = 4 MB
    float* bnd_ws  = base_ws + (size_t)NB * HID;   // NB f32
    float* ndt_ws  = bnd_ws + NB;                  // NB f32

    // Kernel 1: 1024 16-row tiles, one wave per workgroup
    sde_base_gemm<<<dim3(1024), dim3(32), 0, stream>>>(
        z, W1, b1, Wb, bb, Wn, bn, base_ws, bnd_ws, ndt_ws);

    // Kernel 2: one wave per batch element b
    sde_steps<<<dim3(NB), dim3(32), 0, stream>>>(
        W1, W2, b2, noise, base_ws, bnd_ws, ndt_ws, out_scale, out_bias, out);
}
